// GNN_45724221833304
// MI455X (gfx1250) — matrix operand order repaired and checked by reading, not compiled, not run on hardware
//
#include <hip/hip_runtime.h>

// GraphSAGE: S=8 steps, N=50000 nodes, D=128 feats, E=1.6M edges.
// agg = segment_mean(x[src] -> dst); xs = agg@Wl^T + b_l + x@Wr^T; ys = xs@Wfc^T + b_fc

#define S_DIM 8
#define N_DIM 50000
#define D     128
#define E_DIM 1600000

typedef __attribute__((ext_vector_type(16))) __bf16 v16bf;
typedef __attribute__((ext_vector_type(8)))  float  v8f;

union BFrag { uint4 q[2]; v16bf v; };

__device__ __forceinline__ unsigned short f2bf(float f) {
  union { float f; unsigned u; } c; c.f = f;
  unsigned u = c.u + 0x7FFFu + ((c.u >> 16) & 1u);   // round-to-nearest-even
  return (unsigned short)(u >> 16);
}

__global__ void k_zero4(float4* __restrict__ p, long long n4) {
  long long t = (long long)blockIdx.x * blockDim.x + threadIdx.x;
  if (t < n4) p[t] = make_float4(0.f, 0.f, 0.f, 0.f);
}

__global__ void k_w2bf(const float* __restrict__ w, unsigned short* __restrict__ o, int n) {
  int t = blockIdx.x * blockDim.x + threadIdx.x;
  if (t < n) o[t] = f2bf(w[t]);
}

__global__ void k_degree(const int* __restrict__ ei, float* __restrict__ deg) {
  long long t = (long long)blockIdx.x * blockDim.x + threadIdx.x;
  if (t < E_DIM) atomicAdd(&deg[ei[E_DIM + t]], 1.0f);
}

__global__ void k_invdeg(float* __restrict__ deg) {
  int t = blockIdx.x * blockDim.x + threadIdx.x;
  if (t < N_DIM) deg[t] = 1.0f / fmaxf(deg[t], 1.0f);
}

// One thread per (edge, 4-float chunk): float4 gather from x[src], 4 f32 atomics into agg[dst].
// x and agg are L2-resident (25.6 MB each vs 192 MB L2), so this is L2-atomic bound.
__global__ void k_scatter(const float* __restrict__ x, const int* __restrict__ ei,
                          float* __restrict__ agg) {
  long long t = (long long)blockIdx.x * blockDim.x + threadIdx.x;
  long long e = t >> 5;
  if (e >= E_DIM) return;
  int c = (int)(t & 31) * 4;
  long long s = (long long)ei[e];
  long long d = (long long)ei[E_DIM + e];
  const float4 v = *(const float4*)(x + s * D + c);
  float* a = agg + d * D + c;
  atomicAdd(a + 0, v.x);
  atomicAdd(a + 1, v.y);
  atomicAdd(a + 2, v.z);
  atomicAdd(a + 3, v.w);
}

// Block = 256 threads = 8 waves; block handles one 16-node tile, wave w handles
// output columns [16w, 16w+16). D = agg@Wl^T + x@Wr^T + b_l via bf16 WMMA, f32 acc.
__global__ void __launch_bounds__(256) k_gemm(
    const float* __restrict__ x, const float* __restrict__ agg,
    const float* __restrict__ invdeg,
    const unsigned short* __restrict__ Wl, const unsigned short* __restrict__ Wr,
    const float* __restrict__ bl, float* __restrict__ out) {
  __shared__ unsigned short sA[16 * D];  // mean-agg tile, bf16
  __shared__ unsigned short sX[16 * D];  // x tile, bf16
  const long long base = (long long)blockIdx.x * 16;
  const int tid = threadIdx.x;

  for (int i = tid; i < 16 * D; i += 256) {
    int r = i >> 7;
    int k = i & 127;
    long long n = base + r;
    sX[i] = f2bf(x[n * D + k]);
    sA[i] = f2bf(agg[n * D + k] * invdeg[n]);  // fold mean into A
  }
  __syncthreads();

  const int w    = tid >> 5;       // wave id == column tile
  const int lane = tid & 31;
  const int col  = lane & 15;      // A row / B column / D column
  const int hi   = lane >> 4;
  const int kbA  = hi * 8;         // 16-bit A layout k-base per lane half
  const int kbB  = hi * 16;        // unused; B is loaded with kbA, same k order as A
  const int j    = w * 16 + col;   // global output column

  const uint4* wlrow = (const uint4*)(Wl + (long long)j * D);
  const uint4* wrrow = (const uint4*)(Wr + (long long)j * D);

  v8f acc = {};
#pragma unroll
  for (int kc = 0; kc < D; kc += 32) {
    BFrag a, b;
    const uint4* pa = (const uint4*)(sA + col * D + kc + kbA);
    a.q[0] = pa[0];                       // elems 0..7  -> k = kc+kbA+e
    a.q[1] = pa[2];                       // elems 8..15 -> k = kc+kbA+16+(e-8)
    b.q[0] = wlrow[(kc + kbA) >> 3];      // elems 0..7 -> k = kc+kbA+e, 8..15 -> kc+kbA+16+(e-8)
    b.q[1] = wlrow[((kc + kbA) >> 3) + 2];
    acc = __builtin_amdgcn_wmma_f32_16x16x32_bf16(false, a.v, false, b.v,
                                                  (short)0, acc, false, false);
  }
#pragma unroll
  for (int kc = 0; kc < D; kc += 32) {
    BFrag a, b;
    const uint4* pa = (const uint4*)(sX + col * D + kc + kbA);
    a.q[0] = pa[0];
    a.q[1] = pa[2];
    b.q[0] = wrrow[(kc + kbA) >> 3];
    b.q[1] = wrrow[((kc + kbA) >> 3) + 2];
    acc = __builtin_amdgcn_wmma_f32_16x16x32_bf16(false, a.v, false, b.v,
                                                  (short)0, acc, false, false);
  }

  const float bias = bl[j];
#pragma unroll
  for (int r = 0; r < 8; ++r) {
    long long m = base + r + hi * 8;     // C/D layout: VGPR r -> row r (+8 for hi lanes)
    out[m * D + j] = acc[r] + bias;
  }
}

// ys[s,n] = dot(xs[s,n,:], Wfc) + b_fc  — one thread per (s,n), float4 loads.
__global__ void k_fc(const float* __restrict__ xs, const float* __restrict__ wfc,
                     const float* __restrict__ bfc, float* __restrict__ y) {
  long long t = (long long)blockIdx.x * blockDim.x + threadIdx.x;
  if (t >= (long long)S_DIM * N_DIM) return;
  const float4* row = (const float4*)(xs + t * D);
  const float4* wv  = (const float4*)wfc;
  float acc = 0.f;
#pragma unroll
  for (int i = 0; i < D / 4; ++i) {
    float4 a = row[i], b = wv[i];
    acc += a.x * b.x + a.y * b.y + a.z * b.z + a.w * b.w;
  }
  y[t] = acc + bfc[0];
}

extern "C" void kernel_launch(void* const* d_in, const int* in_sizes, int n_in,
                              void* d_out, int out_size, void* d_ws, size_t ws_size,
                              hipStream_t stream) {
  const float* x   = (const float*)d_in[0];   // [S,N,128]
  const int*   ei  = (const int*)d_in[1];     // [2,E]
  const float* Wl  = (const float*)d_in[2];   // [128,128]
  const float* bl  = (const float*)d_in[3];   // [128]
  const float* Wr  = (const float*)d_in[4];   // [128,128]
  const float* Wfc = (const float*)d_in[5];   // [1,128]
  const float* bfc = (const float*)d_in[6];   // [1]

  float* xs = (float*)d_out;                              // [S,N,128]
  float* ys = xs + (long long)S_DIM * N_DIM * D;          // [S,N]

  // workspace layout
  float* invdeg = (float*)d_ws;                           // N floats
  float* agg    = invdeg + N_DIM;                         // N*128 floats
  unsigned short* Wl_bf = (unsigned short*)(agg + (long long)N_DIM * D);
  unsigned short* Wr_bf = Wl_bf + D * D;

  k_w2bf<<<(D * D + 255) / 256, 256, 0, stream>>>(Wl, Wl_bf, D * D);
  k_w2bf<<<(D * D + 255) / 256, 256, 0, stream>>>(Wr, Wr_bf, D * D);

  k_zero4<<<(N_DIM / 4 + 255) / 256, 256, 0, stream>>>((float4*)invdeg, N_DIM / 4);
  k_degree<<<(E_DIM + 255) / 256, 256, 0, stream>>>(ei, invdeg);
  k_invdeg<<<(N_DIM + 255) / 256, 256, 0, stream>>>(invdeg);

  const long long aggN4 = (long long)N_DIM * D / 4;
  for (int s = 0; s < S_DIM; ++s) {
    const float* xstep = x + (long long)s * N_DIM * D;
    k_zero4<<<(int)((aggN4 + 255) / 256), 256, 0, stream>>>((float4*)agg, aggN4);
    long long sth = (long long)E_DIM * 32;
    k_scatter<<<(int)((sth + 255) / 256), 256, 0, stream>>>(xstep, ei, agg);
    k_gemm<<<N_DIM / 16, 256, 0, stream>>>(xstep, agg, invdeg, Wl_bf, Wr_bf, bl,
                                           xs + (long long)s * N_DIM * D);
  }
  k_fc<<<((long long)S_DIM * N_DIM + 255) / 256, 256, 0, stream>>>(xs, Wfc, bfc, ys);
}
